// Attention_1924145349082
// MI455X (gfx1250) — matrix operand order repaired and checked by reading, not compiled, not run on hardware
//
#include <hip/hip_runtime.h>
#include <stdint.h>

typedef __attribute__((ext_vector_type(16))) _Float16 v16h;
typedef __attribute__((ext_vector_type(8)))  _Float16 v8h;
typedef __attribute__((ext_vector_type(4)))  _Float16 v4h;
typedef __attribute__((ext_vector_type(8)))  float    v8f;
typedef __attribute__((ext_vector_type(4)))  unsigned int v4u;
typedef __attribute__((ext_vector_type(8)))  int v8i;
typedef __attribute__((ext_vector_type(4)))  int v4i;

#define NBP  16      // b*p slices
#define NTOK 1024    // tokens per slice
#define DMODEL 512
#define E3   1536    // 3*inner

#define WMMA(a, b, c) \
  __builtin_amdgcn_wmma_f32_16x16x32_f16(false, (a), false, (b), (short)0, (c), false, false)

#if defined(__HIP_DEVICE_COMPILE__) && \
    __has_builtin(__builtin_amdgcn_tensor_load_to_lds) && \
    __has_builtin(__builtin_amdgcn_s_wait_tensorcnt)
#define USE_TDM 1
#else
#define USE_TDM 0
#endif

// A-operand (16x32 f16, M x K) from a row-major matrix: lane(l,g) row = l,
// elements 0..7 -> k = g*8+e, elements 8..15 -> k = 16+g*8+(e-8).
__device__ __forceinline__ v16h load_fragA(const _Float16* rowptr, int g) {
  v8h lo = *(const v8h*)(rowptr + g * 8);
  v8h hi = *(const v8h*)(rowptr + 16 + g * 8);
  v16h r;
#pragma unroll
  for (int e = 0; e < 8; ++e) { r[e] = lo[e]; r[e + 8] = hi[e]; }
  return r;
}

// B-operand (32x16 f16, K x N), column n = row of the [N][K] row-major source:
// lane(l,g) elements 0..7 -> k = g*8+e, elements 8..15 -> k = 16+g*8+(e-8).
__device__ __forceinline__ v16h load_fragB(const _Float16* rowptr, int g) {
  return load_fragA(rowptr, g);
}

#if USE_TDM
__device__ __forceinline__ unsigned lds_off(const void* p) {
  return (unsigned)(uintptr_t)(const __attribute__((address_space(3))) void*)p;
}

// 2D tile load global->LDS via the Tensor Data Mover (D# per ISA 08 §8).
// tile_x elements contiguous (2B each), tile_y rows, row stride stride_x elems.
__device__ __forceinline__ void tdm_load_2d(unsigned ldsaddr, const void* gptr,
                                            unsigned tile_x, unsigned tile_y,
                                            unsigned long long stride_x) {
  unsigned long long ga = (unsigned long long)(uintptr_t)gptr & ((1ull << 57) - 1);
  union { unsigned long long q[2]; v4u v; } G0;
  G0.q[0] = 1ull /*count=1*/ | ((unsigned long long)ldsaddr << 32);
  G0.q[1] = ga | (2ull << 62) /*type=image*/;
  union { unsigned long long q[4]; v8i v; } G1;
  // [17:16]=data_size(2B)=1; [79:48]=tensor_dim0; [111:80]=tensor_dim1;
  // [127:112]=tile_dim0; [143:128]=tile_dim1; [207:160]=tensor_dim0_stride.
  G1.q[0] = 0x10000ull | ((unsigned long long)(tile_x & 0xFFFFu) << 48);
  G1.q[1] = ((unsigned long long)tile_x >> 16) |
            ((unsigned long long)tile_y << 16) |
            ((unsigned long long)tile_x << 48);
  G1.q[2] = (unsigned long long)tile_y | (stride_x << 32);
  G1.q[3] = stride_x >> 32;
  v4i z = {0, 0, 0, 0};
#if __clang_major__ >= 23
  v8i z8 = {0, 0, 0, 0, 0, 0, 0, 0};
  __builtin_amdgcn_tensor_load_to_lds(G0.v, G1.v, z, z, z8, 0);
#else
  __builtin_amdgcn_tensor_load_to_lds(G0.v, G1.v, z, z, 0);
#endif
}
#endif  // USE_TDM

// ---------------- f32 -> f16 conversion ----------------
__global__ void cvt_kernel(const float* __restrict__ src, _Float16* __restrict__ dst, int n) {
  int i = blockIdx.x * blockDim.x + threadIdx.x;
  if (i < n) dst[i] = (_Float16)src[i];
}

// ---------------- LayerNorm (wave per token) ----------------
__global__ void __launch_bounds__(256) ln_kernel(const float* __restrict__ x,
                                                 const float* __restrict__ w,
                                                 const float* __restrict__ b,
                                                 _Float16* __restrict__ Xn) {
  int token = blockIdx.x * 8 + (threadIdx.x >> 5);
  int lane = threadIdx.x & 31;
  const float* row = x + (size_t)token * DMODEL;
  float4 v[4];
  float s = 0.f, s2 = 0.f;
#pragma unroll
  for (int i = 0; i < 4; ++i) {
    v[i] = *(const float4*)(row + (lane + i * 32) * 4);
    s  += v[i].x + v[i].y + v[i].z + v[i].w;
    s2 += v[i].x * v[i].x + v[i].y * v[i].y + v[i].z * v[i].z + v[i].w * v[i].w;
  }
#pragma unroll
  for (int off = 16; off; off >>= 1) {
    s += __shfl_xor(s, off);
    s2 += __shfl_xor(s2, off);
  }
  float mu = s * (1.f / DMODEL);
  float rstd = rsqrtf(s2 * (1.f / DMODEL) - mu * mu + 1e-5f);
  _Float16* orow = Xn + (size_t)token * DMODEL;
#pragma unroll
  for (int i = 0; i < 4; ++i) {
    int c = (lane + i * 32) * 4;
    float4 wv = *(const float4*)(w + c);
    float4 bv = *(const float4*)(b + c);
    v4h o;
    o[0] = (_Float16)((v[i].x - mu) * rstd * wv.x + bv.x);
    o[1] = (_Float16)((v[i].y - mu) * rstd * wv.y + bv.y);
    o[2] = (_Float16)((v[i].z - mu) * rstd * wv.z + bv.z);
    o[3] = (_Float16)((v[i].w - mu) * rstd * wv.w + bv.w);
    *(v4h*)(orow + c) = o;
  }
}

// ---------------- QKV GEMM: qkv = Xn * Wqkv^T (64x64 wave tile) ----------------
// Q,K cols 0..1023 -> QK[bp][tok][1024]; V cols 1024..1535 -> Vt[bp][d][tok] (transposed)
__global__ void __launch_bounds__(128) gemm_qkv_kernel(const _Float16* __restrict__ Xn,
                                                       const _Float16* __restrict__ Wh,
                                                       _Float16* __restrict__ QK,
                                                       _Float16* __restrict__ Vt) {
  const int lane = threadIdx.x & 31;
  const int wave = threadIdx.x >> 5;
  const int l = lane & 15, g = lane >> 4;
  const int bp = blockIdx.z;
  const int m0 = blockIdx.x * 128 + (wave & 1) * 64;
  const int n0 = blockIdx.y * 128 + (wave >> 1) * 64;
  const _Float16* ap = Xn + ((size_t)bp * NTOK + m0 + l) * DMODEL;
  const _Float16* bq = Wh + (size_t)(n0 + l) * DMODEL;
  v8f c[4][4] = {};
  for (int k0 = 0; k0 < DMODEL; k0 += 32) {
    v16h a[4], b[4];
#pragma unroll
    for (int i = 0; i < 4; ++i) a[i] = load_fragA(ap + (size_t)i * 16 * DMODEL + k0, g);
#pragma unroll
    for (int j = 0; j < 4; ++j) b[j] = load_fragB(bq + (size_t)j * 16 * DMODEL + k0, g);
#pragma unroll
    for (int i = 0; i < 4; ++i)
#pragma unroll
      for (int j = 0; j < 4; ++j) c[i][j] = WMMA(a[i], b[j], c[i][j]);
  }
#pragma unroll
  for (int i = 0; i < 4; ++i) {
#pragma unroll
    for (int j = 0; j < 4; ++j) {
      int m = m0 + i * 16 + g * 8;
      int n = n0 + j * 16 + l;
      if (n < 1024) {
        _Float16* p = QK + ((size_t)bp * NTOK + m) * 1024 + n;
#pragma unroll
        for (int r = 0; r < 8; ++r) p[(size_t)r * 1024] = (_Float16)c[i][j][r];
      } else {
        v8h pk;
#pragma unroll
        for (int r = 0; r < 8; ++r) pk[r] = (_Float16)c[i][j][r];
        *(v8h*)(Vt + ((size_t)bp * DMODEL + (n - 1024)) * NTOK + m) = pk;
      }
    }
  }
}

// ---------------- Flash attention (S^T = K*Q^T formulation) ----------------
// Block = 4 waves * 32 queries = 128 queries; K/V tiles (32 keys) staged in LDS
// via TDM async tensor loads (double buffered), shared by all 4 waves.
__global__ void __launch_bounds__(128) attn_kernel(const _Float16* __restrict__ QK,
                                                   const _Float16* __restrict__ Vt,
                                                   _Float16* __restrict__ O) {
  __shared__ __align__(128) _Float16 Ks[2][32][64];  // [buf][key j][dim k]
  __shared__ __align__(128) _Float16 Vs[2][64][32];  // [buf][dim d][key j]
  const int tid = threadIdx.x;
  const int lane = tid & 31, wave = tid >> 5;
  const int l = lane & 15, g = lane >> 4, sb = g * 8;
  const int bp = blockIdx.z, h = blockIdx.y;
  const int i0 = blockIdx.x * 128 + wave * 32;
  const _Float16* base = QK + (size_t)bp * NTOK * 1024;
  const _Float16* kg = base + 512 + h * 64;                      // K rows, ld 1024
  const _Float16* vg = Vt + ((size_t)bp * DMODEL + h * 64) * NTOK;  // V^T rows, ld 1024
  // Q^T B-operand fragments for two 16-query subtiles (persistent)
  v16h qb[2][2];
#pragma unroll
  for (int q = 0; q < 2; ++q) {
    const _Float16* qrow = base + (size_t)(i0 + q * 16 + l) * 1024 + h * 64;
    qb[q][0] = load_fragB(qrow, g);
    qb[q][1] = load_fragB(qrow + 32, g);
  }
  v8f o[2][4] = {};
  float Mi[2] = {-1e30f, -1e30f}, Li[2] = {0.f, 0.f};
  const float scale = 0.125f;  // 64^-0.5

#if USE_TDM
  if (wave == 0) {
    tdm_load_2d(lds_off(&Ks[0][0][0]), kg, 64, 32, 1024ull);
    tdm_load_2d(lds_off(&Vs[0][0][0]), vg, 32, 64, 1024ull);
  }
#endif
  for (int t = 0; t < NTOK / 32; ++t) {
    const int j0 = t * 32;
    const int buf = t & 1;
#if USE_TDM
    if (wave == 0) {
      if (t + 1 < NTOK / 32) {  // prefetch next tile into other buffer
        tdm_load_2d(lds_off(&Ks[buf ^ 1][0][0]), kg + (size_t)(j0 + 32) * 1024, 64, 32, 1024ull);
        tdm_load_2d(lds_off(&Vs[buf ^ 1][0][0]), vg + j0 + 32, 32, 64, 1024ull);
        __builtin_amdgcn_s_wait_tensorcnt(2);  // current tile (2 older ops) complete
      } else {
        __builtin_amdgcn_s_wait_tensorcnt(0);
      }
    }
    __syncthreads();  // tile t visible to all waves
#else
    {  // cooperative fallback copy: 32B per thread per tile
      int kr = tid >> 2, kc = (tid & 3) * 16;
      *(v16h*)&Ks[buf][kr][kc] = *(const v16h*)(kg + (size_t)(j0 + kr) * 1024 + kc);
      int vr = tid >> 1, vc = (tid & 1) * 16;
      *(v16h*)&Vs[buf][vr][vc] = *(const v16h*)(vg + (size_t)vr * NTOK + j0 + vc);
    }
    __syncthreads();
#endif
    // K A-operand frags and V B-operand frags from LDS (shared by all waves)
    v16h ka[2][2];
#pragma unroll
    for (int s = 0; s < 2; ++s) {
      ka[s][0] = load_fragA(&Ks[buf][s * 16 + l][0], g);
      ka[s][1] = load_fragA(&Ks[buf][s * 16 + l][32], g);
    }
    v16h vb[4];
#pragma unroll
    for (int dt = 0; dt < 4; ++dt) vb[dt] = load_fragB(&Vs[buf][dt * 16 + l][0], g);

#pragma unroll
    for (int q = 0; q < 2; ++q) {
      v8f st0 = {}, st1 = {};
      st0 = WMMA(ka[0][0], qb[q][0], st0);  // S^T[j0..j0+15][i = i0+q*16+l]
      st0 = WMMA(ka[0][1], qb[q][1], st0);
      st1 = WMMA(ka[1][0], qb[q][0], st1);  // S^T[j0+16..j0+31][...]
      st1 = WMMA(ka[1][1], qb[q][1], st1);
      float tm = -1e30f;
#pragma unroll
      for (int r = 0; r < 8; ++r) {
        st0[r] *= scale; st1[r] *= scale;
        tm = fmaxf(tm, fmaxf(st0[r], st1[r]));
      }
      tm = fmaxf(tm, __shfl_xor(tm, 16));  // combine the two j-halves of query l
      float mn = fmaxf(Mi[q], tm);
      float alpha = __expf(Mi[q] - mn);
      float rs = 0.f;
      v16h pa;  // exp(S^T) accum frags ARE the A-operand of P*V (layout matches)
#pragma unroll
      for (int r = 0; r < 8; ++r) {
        float e0 = __expf(st0[r] - mn);
        float e1 = __expf(st1[r] - mn);
        rs += e0 + e1;
        pa[r] = (_Float16)e0;
        pa[r + 8] = (_Float16)e1;
      }
      rs += __shfl_xor(rs, 16);
      Li[q] = Li[q] * alpha + rs;
      Mi[q] = mn;
#pragma unroll
      for (int r = 0; r < 8; ++r) {  // O row i = r+g*8 -> alpha lives in lane sb+r
        float ar = __shfl(alpha, sb + r);
        o[q][0][r] *= ar; o[q][1][r] *= ar; o[q][2][r] *= ar; o[q][3][r] *= ar;
      }
#pragma unroll
      for (int dt = 0; dt < 4; ++dt) o[q][dt] = WMMA(pa, vb[dt], o[q][dt]);
    }
    __syncthreads();  // all waves done reading buf before it is refilled
  }
#pragma unroll
  for (int q = 0; q < 2; ++q) {
    float linv = 1.f / Li[q];
    _Float16* obase = O + ((size_t)bp * NTOK + i0 + q * 16 + sb) * DMODEL + h * 64;
#pragma unroll
    for (int r = 0; r < 8; ++r) {
      float lr = __shfl(linv, sb + r);
      _Float16* p = obase + (size_t)r * DMODEL;
      p[0 + l]  = (_Float16)(o[q][0][r] * lr);
      p[16 + l] = (_Float16)(o[q][1][r] * lr);
      p[32 + l] = (_Float16)(o[q][2][r] * lr);
      p[48 + l] = (_Float16)(o[q][3][r] * lr);
    }
  }
}

// ---------------- Output GEMM: out = O * Wout^T (fp32 result, 64x64 wave tile) ----------------
__global__ void __launch_bounds__(128) gemm_out_kernel(const _Float16* __restrict__ O,
                                                       const _Float16* __restrict__ Wh,
                                                       float* __restrict__ out) {
  const int lane = threadIdx.x & 31;
  const int wave = threadIdx.x >> 5;
  const int l = lane & 15, g = lane >> 4;
  const int bp = blockIdx.z;
  const int m0 = blockIdx.x * 128 + (wave & 1) * 64;
  const int n0 = blockIdx.y * 128 + (wave >> 1) * 64;
  const _Float16* ap = O + ((size_t)bp * NTOK + m0 + l) * DMODEL;
  const _Float16* bq = Wh + (size_t)(n0 + l) * DMODEL;
  v8f c[4][4] = {};
  for (int k0 = 0; k0 < DMODEL; k0 += 32) {
    v16h a[4], b[4];
#pragma unroll
    for (int i = 0; i < 4; ++i) a[i] = load_fragA(ap + (size_t)i * 16 * DMODEL + k0, g);
#pragma unroll
    for (int j = 0; j < 4; ++j) b[j] = load_fragB(bq + (size_t)j * 16 * DMODEL + k0, g);
#pragma unroll
    for (int i = 0; i < 4; ++i)
#pragma unroll
      for (int j = 0; j < 4; ++j) c[i][j] = WMMA(a[i], b[j], c[i][j]);
  }
#pragma unroll
  for (int i = 0; i < 4; ++i) {
#pragma unroll
    for (int j = 0; j < 4; ++j) {
      float* p = out + ((size_t)bp * NTOK + m0 + i * 16 + g * 8) * DMODEL + n0 + j * 16 + l;
#pragma unroll
      for (int r = 0; r < 8; ++r) p[(size_t)r * DMODEL] = c[i][j][r];
    }
  }
}

extern "C" void kernel_launch(void* const* d_in, const int* in_sizes, int n_in,
                              void* d_out, int out_size, void* d_ws, size_t ws_size,
                              hipStream_t stream) {
  const float* x     = (const float*)d_in[0];
  const float* ln_w  = (const float*)d_in[1];
  const float* ln_b  = (const float*)d_in[2];
  const float* w_qkv = (const float*)d_in[3];
  const float* w_out = (const float*)d_in[4];
  float* out = (float*)d_out;

  char* ws = (char*)d_ws;
  size_t off = 0;
  _Float16* Xn = (_Float16*)(ws + off); off += (size_t)NBP * NTOK * DMODEL * 2;   // 16 MB
  _Float16* QK = (_Float16*)(ws + off); off += (size_t)NBP * NTOK * 1024 * 2;     // 32 MB
  _Float16* Vt = (_Float16*)(ws + off); off += (size_t)NBP * DMODEL * NTOK * 2;   // 16 MB
  _Float16* Ob = (_Float16*)(ws + off); off += (size_t)NBP * NTOK * DMODEL * 2;   // 16 MB
  _Float16* WqkvH = (_Float16*)(ws + off); off += (size_t)E3 * DMODEL * 2;        // 1.5 MB
  _Float16* WoutH = (_Float16*)(ws + off); off += (size_t)DMODEL * DMODEL * 2;    // 0.5 MB

  const int nq = E3 * DMODEL;       // 786432
  const int no = DMODEL * DMODEL;   // 262144
  cvt_kernel<<<(nq + 255) / 256, 256, 0, stream>>>(w_qkv, WqkvH, nq);
  cvt_kernel<<<(no + 255) / 256, 256, 0, stream>>>(w_out, WoutH, no);
  ln_kernel<<<(NBP * NTOK) / 8, 256, 0, stream>>>(x, ln_w, ln_b, Xn);
  gemm_qkv_kernel<<<dim3(8, 12, NBP), 128, 0, stream>>>(Xn, WqkvH, QK, Vt);
  attn_kernel<<<dim3(8, 8, NBP), 128, 0, stream>>>(QK, Vt, Ob);
  gemm_out_kernel<<<dim3(8, 4, NBP), 128, 0, stream>>>(Ob, WoutH, out);
}
